// PairwiseScore_19430432047650
// MI455X (gfx1250) — matrix operand order repaired and checked by reading, not compiled, not run on hardware
//
#include <hip/hip_runtime.h>
#include <hip/hip_bf16.h>

// ---------------- problem constants ----------------
#define GI      1220          // g_i feature dim
#define HIDN    150           // hidden size
#define NPAD    160           // hidden padded to 10 WMMA n-tiles
#define NT      10            // n-tiles (one per wave)
#define CW      128           // chunk width over original g-features
#define NCHG    10            // chunks covering 1220 (padded to 1280)
#define CH_W    (3*CW)        // 384 A-columns per chunk (i, j, i*j)
#define KST_CH  (CH_W/32)     // 12 wmma k-steps per chunk
#define PHI_W   64            // phi(60) padded to 64
#define KST_PHI (PHI_W/32)    // 2
#define KST_L1  (NCHG*KST_CH + KST_PHI)   // 122
#define KST_L2  (NPAD/32)                 // 5
#define W1FRAG_UINTS (KST_L1*NT*32*8)     // 312320 uints (~1.25 MB)
#define W2FRAG_UINTS (KST_L2*NT*32*8)     // 12800 uints

typedef __attribute__((ext_vector_type(16))) __bf16 v16bf;
typedef __attribute__((ext_vector_type(8)))  float  v8f;

union FragU { uint4 q[2]; v16bf v; };

__device__ __forceinline__ unsigned short f2bf(float x) {
  unsigned u = __float_as_uint(x);
  return (unsigned short)((u + 0x7FFFu + ((u >> 16) & 1u)) >> 16); // RNE
}
__device__ __forceinline__ float bf2f(unsigned short h) {
  return __uint_as_float(((unsigned)h) << 16);
}

// padded-K position -> original K in [0,3720), or -1 for zero padding
__device__ __forceinline__ int map_k1(int kp) {
  if (kp < NCHG * CH_W) {
    int c  = kp / CH_W;
    int t  = kp - c * CH_W;
    int r  = t / CW;                  // 0:i  1:j  2:i*j
    int kk = c * CW + (t - r * CW);   // original feature index
    return (kk < GI) ? r * GI + kk : -1;
  }
  int t = kp - NCHG * CH_W;           // phi region
  return (t < 60) ? (3 * GI + t) : -1;
}

// ---------------- weight repack (bf16, WMMA-fragment layout) ----------------
// frag[((kstep*NT + nt)*32 + lane)*8 + v] packs B[k0][n],B[k1][n]
// with n = nt*16 + (lane&15), k0 = kstep*32 + (lane>>4)*16 + 2v, k1 = k0+1.
__global__ void build_w1frag(const float* __restrict__ W1,
                             unsigned int* __restrict__ frag) {
  for (int i = blockIdx.x * blockDim.x + threadIdx.x; i < W1FRAG_UINTS;
       i += gridDim.x * blockDim.x) {
    int v    = i & 7;
    int lane = (i >> 3) & 31;
    int g    = i >> 8;                 // kstep*NT + nt
    int nt = g % NT, kstep = g / NT;
    int n   = nt * 16 + (lane & 15);
    int kp0 = kstep * 32 + (lane >> 4) * 16 + 2 * v;
    unsigned short h0 = 0, h1 = 0;
    if (n < HIDN) {
      int k0 = map_k1(kp0), k1 = map_k1(kp0 + 1);
      if (k0 >= 0) h0 = f2bf(W1[(long)k0 * HIDN + n]);
      if (k1 >= 0) h1 = f2bf(W1[(long)k1 * HIDN + n]);
    }
    frag[i] = (unsigned)h0 | ((unsigned)h1 << 16);
  }
}

__global__ void build_w2frag(const float* __restrict__ W2,
                             unsigned int* __restrict__ frag) {
  for (int i = blockIdx.x * blockDim.x + threadIdx.x; i < W2FRAG_UINTS;
       i += gridDim.x * blockDim.x) {
    int v    = i & 7;
    int lane = (i >> 3) & 31;
    int g    = i >> 8;
    int nt = g % NT, kstep = g / NT;
    int n   = nt * 16 + (lane & 15);
    int kp0 = kstep * 32 + (lane >> 4) * 16 + 2 * v;
    unsigned short h0 = 0, h1 = 0;
    if (n < HIDN) {
      if (kp0     < HIDN) h0 = f2bf(W2[(long)kp0 * HIDN + n]);
      if (kp0 + 1 < HIDN) h1 = f2bf(W2[(long)(kp0 + 1) * HIDN + n]);
    }
    frag[i] = (unsigned)h0 | ((unsigned)h1 << 16);
  }
}

// ---------------- WMMA inner loop ----------------
// A fragment (contiguous K order, matching B frag): lanes 0-15 hold K 0-15,
// lanes 16-31 hold K 16-31 -> two 16B contiguous LDS reads.
__device__ __forceinline__ void mma_chunk(const unsigned short* A, int rstride,
                                          int nsteps, int kstep0,
                                          const unsigned int* __restrict__ Wfrag,
                                          int nt, int lane, v8f& acc) {
  const int half = lane >> 4;
  const unsigned short* arow = A + (lane & 15) * rstride;
  for (int s = 0; s < nsteps; ++s) {
    FragU a, b;
    const unsigned short* ap = arow + s * 32 + half * 16;
    a.q[0] = *(const uint4*)(ap);        // ds_load_b128
    a.q[1] = *(const uint4*)(ap + 8);    // ds_load_b128
    const uint4* bp =
        (const uint4*)(Wfrag + (size_t)((((kstep0 + s) * NT + nt) * 32 + lane) * 8));
    b.q[0] = bp[0];                      // global_load_b128
    b.q[1] = bp[1];                      // global_load_b128
    acc = __builtin_amdgcn_wmma_f32_16x16x32_bf16(
        false, a.v, false, b.v, (short)0, acc, false, false);
  }
}

// ---------------- main fused kernel: 16 pairs per workgroup ----------------
__global__ __launch_bounds__(320) void pairwise_mlp(
    const float* __restrict__ g_i, const float* __restrict__ mscore,
    const float* __restrict__ demb, const float* __restrict__ gemb,
    const float* __restrict__ semb,
    const float* __restrict__ b1, const float* __restrict__ b2,
    const float* __restrict__ W3, const float* __restrict__ b3,
    const int* __restrict__ mid, const int* __restrict__ aid,
    const int* __restrict__ did, const int* __restrict__ gid,
    const int* __restrict__ sid,
    const unsigned int* __restrict__ W1frag,
    const unsigned int* __restrict__ W2frag,
    float* __restrict__ out, int P) {
  __shared__ __attribute__((aligned(32))) unsigned short A_lds[16 * CH_W];
  __shared__ __attribute__((aligned(32))) unsigned short h1_lds[16 * NPAD];
  __shared__ __attribute__((aligned(32))) unsigned short h2_lds[16 * NPAD];
  __shared__ int s_mi[16], s_ai[16], s_di[16], s_gi[16], s_si[16];
  __shared__ float s_ms[16];

  const int tid  = threadIdx.x;
  const int lane = tid & 31;
  const int nt   = tid >> 5;      // wave id == n-tile
  const int half = lane >> 4;
  const int p0   = blockIdx.x * 16;

  if (tid < 16) {
    int p = min(p0 + tid, P - 1);
    int mi = mid[p], ai = aid[p];
    s_mi[tid] = mi; s_ai[tid] = ai;
    s_di[tid] = did[p]; s_gi[tid] = gid[p]; s_si[tid] = sid[p];
    s_ms[tid] = mscore[mi] + mscore[ai];
  }

  v8f acc = {0.f, 0.f, 0.f, 0.f, 0.f, 0.f, 0.f, 0.f};

  // ---- layer 1: K reordered so each g_i read feeds i, j and i*j columns ----
  for (int c = 0; c < NCHG; ++c) {
    __syncthreads();                       // previous chunk fully consumed
    const int kk0 = c * CW;
    for (int idx = tid; idx < 16 * CW; idx += 320) {
      int row = idx >> 7, col = idx & (CW - 1);
      int kk = kk0 + col;
      float iv = 0.f, jv = 0.f;
      if (kk < GI) {
        iv = g_i[(long)s_mi[row] * GI + kk];
        jv = g_i[(long)s_ai[row] * GI + kk];
      }
      unsigned short* r = A_lds + row * CH_W;
      r[col]          = f2bf(iv);
      r[CW + col]     = f2bf(jv);
      r[2 * CW + col] = f2bf(iv * jv);
    }
    __syncthreads();
    mma_chunk(A_lds, CH_W, KST_CH, c * KST_CH, W1frag, nt, lane, acc);
  }

  // phi chunk (dist/genre/speaker embeddings, padded to 64)
  __syncthreads();
  for (int idx = tid; idx < 16 * PHI_W; idx += 320) {
    int row = idx >> 6, col = idx & (PHI_W - 1);
    float v = 0.f;
    if      (col < 20) v = demb[s_di[row] * 20 + col];
    else if (col < 40) v = gemb[s_gi[row] * 20 + col - 20];
    else if (col < 60) v = semb[s_si[row] * 20 + col - 40];
    A_lds[row * PHI_W + col] = f2bf(v);
  }
  __syncthreads();
  mma_chunk(A_lds, PHI_W, KST_PHI, NCHG * KST_CH, W1frag, nt, lane, acc);

  // bias + relu -> h1 (bf16 in LDS); padded columns stay exactly 0
  {
    int n = nt * 16 + (lane & 15);
    float bv = (n < HIDN) ? b1[n] : 0.f;
#pragma unroll
    for (int r = 0; r < 8; ++r) {
      float x = acc[r] + bv;
      h1_lds[(half * 8 + r) * NPAD + n] = f2bf(fmaxf(x, 0.f));
    }
  }
  __syncthreads();

  // ---- layer 2 ----
  v8f acc2 = {0.f, 0.f, 0.f, 0.f, 0.f, 0.f, 0.f, 0.f};
  mma_chunk(h1_lds, NPAD, KST_L2, 0, W2frag, nt, lane, acc2);
  {
    int n = nt * 16 + (lane & 15);
    float bv = (n < HIDN) ? b2[n] : 0.f;
#pragma unroll
    for (int r = 0; r < 8; ++r) {
      float x = acc2[r] + bv;
      h2_lds[(half * 8 + r) * NPAD + n] = f2bf(fmaxf(x, 0.f));
    }
  }
  __syncthreads();

  // ---- layer 3: 150-dot + scores (negligible work, wave 0) ----
  if (tid < 16) {
    float sum = b3[0];
    const unsigned short* hr = h2_lds + tid * NPAD;
    for (int k = 0; k < HIDN; ++k) sum += bf2f(hr[k]) * W3[k];
    int p = p0 + tid;
    if (p < P) out[p] = s_ms[tid] + sum;
  }
}

// ---------------- launch ----------------
extern "C" void kernel_launch(void* const* d_in, const int* in_sizes, int n_in,
                              void* d_out, int out_size, void* d_ws, size_t ws_size,
                              hipStream_t stream) {
  const float* g_i  = (const float*)d_in[0];
  const float* ms   = (const float*)d_in[1];
  const float* demb = (const float*)d_in[2];
  const float* gemb = (const float*)d_in[3];
  const float* semb = (const float*)d_in[4];
  const float* W1   = (const float*)d_in[5];
  const float* b1   = (const float*)d_in[6];
  const float* W2   = (const float*)d_in[7];
  const float* b2   = (const float*)d_in[8];
  const float* W3   = (const float*)d_in[9];
  const float* b3   = (const float*)d_in[10];
  const int* mid = (const int*)d_in[11];
  const int* aid = (const int*)d_in[12];
  const int* did = (const int*)d_in[13];
  const int* gid = (const int*)d_in[14];
  const int* sid = (const int*)d_in[15];
  const int P = in_sizes[11];

  unsigned int* W1frag = (unsigned int*)d_ws;
  unsigned int* W2frag = W1frag + W1FRAG_UINTS;

  build_w1frag<<<512, 256, 0, stream>>>(W1, W1frag);
  build_w2frag<<<50, 256, 0, stream>>>(W2, W2frag);

  int nblk = (P + 15) / 16;
  pairwise_mlp<<<nblk, 320, 0, stream>>>(g_i, ms, demb, gemb, semb,
                                         b1, b2, W3, b3,
                                         mid, aid, did, gid, sid,
                                         W1frag, W2frag, (float*)d_out, P);
}
